// GIN_16870631539207
// MI455X (gfx1250) — matrix operand order repaired and checked by reading, not compiled, not run on hardware
//
#include <hip/hip_runtime.h>
#include <hip/hip_bf16.h>

typedef __attribute__((ext_vector_type(16))) __bf16 v16bf;
typedef __attribute__((ext_vector_type(8)))  float  v8f;

#define GN 50000
#define GE 400000
#define DIN 771
#define HID 256
#define NGRAPH 512

// ---------------------------------------------------------------- utility
__global__ void zero_f32(float* __restrict__ p, int n) {
  int i = blockIdx.x * blockDim.x + threadIdx.x;
  if (i < n) p[i] = 0.0f;
}

// z[i] = (1 + eps) * hin[i]
__global__ void init_z(float* __restrict__ z, const float* __restrict__ hin,
                       const float* __restrict__ epsp, int n) {
  int i = blockIdx.x * blockDim.x + threadIdx.x;
  if (i < n) z[i] = (1.0f + epsp[0]) * hin[i];
}

// wave-per-edge scatter add: z[dst] += h[src]
__global__ void scatter_edges(float* __restrict__ z, const float* __restrict__ h,
                              const int* __restrict__ src, const int* __restrict__ dst,
                              int E, int d) {
  int wid  = (blockIdx.x * blockDim.x + threadIdx.x) >> 5;
  int lane = threadIdx.x & 31;
  if (wid >= E) return;
  int s = src[wid];
  int t = dst[wid];
  const float* hs = h + (size_t)s * d;
  float*       zt = z + (size_t)t * d;
  for (int f = lane; f < d; f += 32) atomicAdd(&zt[f], hs[f]);
}

// ---------------------------------------------------------------- WMMA GEMM
// C[M,256] = A[M,K] (f32) @ W[K,256] (f32) + bias, via bf16 WMMA, f32 accum.
// Block: 256 threads (8 waves), tile BM=64 x BN=256, BK=32.
// Double-buffered LDS with register prefetch: one barrier per K-step.
__global__ __launch_bounds__(256) void gemm_bf16_wmma(
    const float* __restrict__ A, const float* __restrict__ W,
    const float* __restrict__ bias, float* __restrict__ C, int M, int K) {
  __shared__ __bf16 As[2][64][40];   // 64 x 32 (+8 pad, 80B rows -> b128 frags)
  __shared__ __bf16 Bs[2][256][40];  // 256 cols x 32 K (+8 pad), transposed

  const int tid  = threadIdx.x;
  const int lane = tid & 31;
  const int w    = tid >> 5;   // wave 0..7
  const int rt   = w & 3;      // row tile within block (16 rows each)
  const int cg   = w >> 2;     // column half: 0 -> cols 0..127, 1 -> 128..255
  const int row0 = blockIdx.x * 64;

  const int m_fr = lane & 15;
  const int hi   = lane >> 4;
  const int kb   = hi * 8;     // A-frag K base
  const int kb2  = hi * 16;    // unused; B-frag also uses kb

  // staging assignment
  const int ar  = tid >> 2;         // A row 0..63
  const int ac0 = (tid & 3) * 8;    // A col start {0,8,16,24}
  const int bn  = tid;              // B column 0..255 (column-per-thread)

  v8f acc[8] = {};
  float aReg[8];
  float bReg[32];

  auto fetch = [&](int k0) {
    const int gr = row0 + ar;
    const float* arow = A + (size_t)gr * K + k0 + ac0;
#pragma unroll
    for (int j = 0; j < 8; ++j) {
      int kk = k0 + ac0 + j;
      aReg[j] = (gr < M && kk < K) ? arow[j] : 0.0f;
    }
    // column bn of W: at each k the whole block reads one contiguous W row
    const float* wcol = W + (size_t)k0 * 256 + bn;
#pragma unroll
    for (int j = 0; j < 32; ++j) bReg[j] = (k0 + j < K) ? wcol[(size_t)j * 256] : 0.0f;
  };
  auto stage = [&](int p) {
#pragma unroll
    for (int j = 0; j < 8; ++j) As[p][ar][ac0 + j] = (__bf16)aReg[j];
    // 32 consecutive bf16 in one 16B-aligned LDS row -> 4x ds_store_b128
#pragma unroll
    for (int j = 0; j < 32; ++j) Bs[p][bn][j] = (__bf16)bReg[j];
  };

  fetch(0);
  stage(0);

  int p = 0;
  for (int k0 = 0; k0 < K; k0 += 32) {
    const bool more = (k0 + 32) < K;
    if (more) fetch(k0 + 32);      // prefetch next tile into registers
    __syncthreads();               // stage(p) from previous step now visible

    // A fragment: lane holds row m, K halves [kb..kb+7] and [16+kb..16+kb+7]
    v16bf a;
    {
      int m = rt * 16 + m_fr;
#pragma unroll
      for (int j = 0; j < 8; ++j) {
        a[j]     = As[p][m][kb + j];
        a[j + 8] = As[p][m][16 + kb + j];
      }
    }
#pragma unroll
    for (int ct = 0; ct < 8; ++ct) {
      int n = cg * 128 + ct * 16 + m_fr;
      v16bf b;
#pragma unroll
      for (int j = 0; j < 16; ++j) b[j] = Bs[p][n][(j < 8) ? (kb + j) : (8 + kb + j)];
      acc[ct] = __builtin_amdgcn_wmma_f32_16x16x32_bf16(
          false, a, false, b, (short)0, acc[ct], false, false);
    }

    if (more) stage(p ^ 1);        // write next tile into the other buffer
    p ^= 1;
  }

  // epilogue: D layout -> m = rt*16 + r + 8*hi, n = cg*128 + ct*16 + (lane&15)
  const int mbase = rt * 16 + hi * 8;
#pragma unroll
  for (int ct = 0; ct < 8; ++ct) {
    int n = cg * 128 + ct * 16 + m_fr;
    float bv = bias[n];
#pragma unroll
    for (int r = 0; r < 8; ++r) {
      int gr = row0 + mbase + r;
      if (gr < M) C[(size_t)gr * 256 + n] = acc[ct][r] + bv;
    }
  }
}

// ---------------------------------------------------------------- BatchNorm
// per-column sum / sumsq over M rows (256 columns)
__global__ void bn_stats(const float* __restrict__ t, float* __restrict__ stats, int M) {
  int c    = threadIdx.x;          // 256 columns
  int row0 = blockIdx.x * 128;
  float s = 0.0f, q = 0.0f;
  int rend = row0 + 128;
  if (rend > M) rend = M;
  for (int r = row0; r < rend; ++r) {
    float x = t[(size_t)r * 256 + c];
    s += x;
    q += x * x;
  }
  atomicAdd(&stats[c], s);
  atomicAdd(&stats[256 + c], q);
}

// stats -> in-place (scale, shift)
__global__ void bn_finalize(float* __restrict__ stats, const float* __restrict__ g,
                            const float* __restrict__ be, int M) {
  int c = threadIdx.x;
  float mu  = stats[c] / (float)M;
  float var = stats[256 + c] / (float)M - mu * mu;
  float sc  = g[c] * rsqrtf(var + 1e-5f);
  stats[c]       = sc;
  stats[256 + c] = be[c] - mu * sc;
}

__global__ void bn_apply_relu(float* __restrict__ out, const float* __restrict__ t,
                              const float* __restrict__ stats, int n) {
  int i = blockIdx.x * blockDim.x + threadIdx.x;
  if (i >= n) return;
  int c = i & 255;
  float y = fmaf(t[i], stats[c], stats[256 + c]);
  out[i] = fmaxf(y, 0.0f);
}

// ---------------------------------------------------------------- pooling
// wave-per-node: pooled[batch[n]*768 + off + c] += h[n][c]
__global__ void pool_add(float* __restrict__ pooled, const float* __restrict__ h,
                         const int* __restrict__ batch, int Nn, int off) {
  int wid  = (blockIdx.x * blockDim.x + threadIdx.x) >> 5;
  int lane = threadIdx.x & 31;
  if (wid >= Nn) return;
  int g = batch[wid];
  const float* hr = h + (size_t)wid * 256;
  float* pr = pooled + (size_t)g * 768 + off;
  for (int c = lane; c < 256; c += 32) atomicAdd(&pr[c], hr[c]);
}

// ---------------------------------------------------------------- classifier
__global__ __launch_bounds__(256) void clf_kernel(
    const float* __restrict__ J, const float* __restrict__ W1, const float* __restrict__ b1,
    const float* __restrict__ W2, const float* __restrict__ b2,
    const float* __restrict__ W3, const float* __restrict__ b3, float* __restrict__ out) {
  __shared__ float row[768];
  __shared__ float y1[256];
  __shared__ float y2[128];
  __shared__ float red[256];
  int g = blockIdx.x, t = threadIdx.x;
  for (int i = t; i < 768; i += 256) row[i] = J[(size_t)g * 768 + i];
  __syncthreads();
  {
    float a = b1[t];
    for (int k = 0; k < 768; ++k) a = fmaf(row[k], W1[(size_t)k * 256 + t], a);
    y1[t] = fmaxf(a, 0.0f);
  }
  __syncthreads();
  if (t < 128) {
    float a = b2[t];
    for (int k = 0; k < 256; ++k) a = fmaf(y1[k], W2[(size_t)k * 128 + t], a);
    y2[t] = fmaxf(a, 0.0f);
  }
  __syncthreads();
  red[t] = (t < 128) ? y2[t] * W3[t] : 0.0f;
  __syncthreads();
  for (int s = 128; s > 0; s >>= 1) {
    if (t < s) red[t] += red[t + s];
    __syncthreads();
  }
  if (t == 0) out[g] = red[0] + b3[0];
}

// ---------------------------------------------------------------- launch
extern "C" void kernel_launch(void* const* d_in, const int* in_sizes, int n_in,
                              void* d_out, int out_size, void* d_ws, size_t ws_size,
                              hipStream_t stream) {
  (void)in_sizes; (void)n_in; (void)out_size; (void)ws_size;

  const float* x = (const float*)d_in[0];
  // params: l0..l2 at 1+9*i: W1,b1,g1,be1,W2,b2,g2,be2,eps ; clf at 28..33
  const int* ei    = (const int*)d_in[34];
  const int* batch = (const int*)d_in[35];
  const int* src = ei;
  const int* dst = ei + GE;

  char* ws = (char*)d_ws;
  size_t off = 0;
  auto alloc = [&](size_t bytes) -> float* {
    off = (off + 255) & ~(size_t)255;
    float* p = (float*)(ws + off);
    off += bytes;
    return p;
  };
  float* z      = alloc((size_t)GN * DIN * 4);   // aggregation buffer (max width)
  float* t      = alloc((size_t)GN * HID * 4);   // GEMM output / reuse
  float* u      = alloc((size_t)GN * HID * 4);   // BN1 output (GEMM2 input)
  float* h      = alloc((size_t)GN * HID * 4);   // layer output
  float* stats  = alloc(512 * 4);
  float* pooled = alloc((size_t)NGRAPH * 768 * 4);

  const int TB = 256;
  dim3 blk(TB);

  zero_f32<<<dim3((NGRAPH * 768 + TB - 1) / TB), blk, 0, stream>>>(pooled, NGRAPH * 768);

  const int gemm_blocks = (GN + 63) / 64;
  const int stat_blocks = (GN + 127) / 128;
  const int edge_waves  = (GE + 7) / 8;   // 8 waves per block
  const int node_waves  = (GN + 7) / 8;

  for (int i = 0; i < 3; ++i) {
    const float* W1  = (const float*)d_in[1 + 9 * i + 0];
    const float* b1  = (const float*)d_in[1 + 9 * i + 1];
    const float* g1  = (const float*)d_in[1 + 9 * i + 2];
    const float* be1 = (const float*)d_in[1 + 9 * i + 3];
    const float* W2  = (const float*)d_in[1 + 9 * i + 4];
    const float* b2  = (const float*)d_in[1 + 9 * i + 5];
    const float* g2  = (const float*)d_in[1 + 9 * i + 6];
    const float* be2 = (const float*)d_in[1 + 9 * i + 7];
    const float* eps = (const float*)d_in[1 + 9 * i + 8];

    const float* hin = (i == 0) ? x : h;
    const int d = (i == 0) ? DIN : HID;
    const int nz = GN * d;

    // z = (1+eps)*h ; z[dst] += h[src]
    init_z<<<dim3((nz + TB - 1) / TB), blk, 0, stream>>>(z, hin, eps, nz);
    scatter_edges<<<dim3(edge_waves), blk, 0, stream>>>(z, hin, src, dst, GE, d);

    // t = z @ W1 + b1
    gemm_bf16_wmma<<<dim3(gemm_blocks), blk, 0, stream>>>(z, W1, b1, t, GN, d);

    // BN1 + ReLU -> u
    zero_f32<<<dim3(2), blk, 0, stream>>>(stats, 512);
    bn_stats<<<dim3(stat_blocks), blk, 0, stream>>>(t, stats, GN);
    bn_finalize<<<dim3(1), blk, 0, stream>>>(stats, g1, be1, GN);
    bn_apply_relu<<<dim3((GN * HID + TB - 1) / TB), blk, 0, stream>>>(u, t, stats, GN * HID);

    // t = u @ W2 + b2
    gemm_bf16_wmma<<<dim3(gemm_blocks), blk, 0, stream>>>(u, W2, b2, t, GN, HID);

    // BN2 + ReLU -> h
    zero_f32<<<dim3(2), blk, 0, stream>>>(stats, 512);
    bn_stats<<<dim3(stat_blocks), blk, 0, stream>>>(t, stats, GN);
    bn_finalize<<<dim3(1), blk, 0, stream>>>(stats, g2, be2, GN);
    bn_apply_relu<<<dim3((GN * HID + TB - 1) / TB), blk, 0, stream>>>(h, t, stats, GN * HID);

    // jumping-knowledge pooling
    pool_add<<<dim3(node_waves), blk, 0, stream>>>(pooled, h, batch, GN, i * HID);
  }

  const float* cW1 = (const float*)d_in[28];
  const float* cb1 = (const float*)d_in[29];
  const float* cW2 = (const float*)d_in[30];
  const float* cb2 = (const float*)d_in[31];
  const float* cW3 = (const float*)d_in[32];
  const float* cb3 = (const float*)d_in[33];
  clf_kernel<<<dim3(NGRAPH), blk, 0, stream>>>(pooled, cW1, cb1, cW2, cb2, cW3, cb3,
                                               (float*)d_out);
}
